// ARNet_4698694221996
// MI455X (gfx1250) — matrix operand order repaired and checked by reading, not compiled, not run on hardware
//
#include <hip/hip_runtime.h>

typedef _Float16 f16_t;
typedef __attribute__((ext_vector_type(16))) _Float16 v16h;
typedef __attribute__((ext_vector_type(8)))  _Float16 v8h;
typedef __attribute__((ext_vector_type(8)))  float    v8f;
typedef __attribute__((ext_vector_type(4)))  float    v4f;

union Frag16 { v16h v; v8h h[2]; };

#define U_PITCH 40  // 32 cols + 8 pad halfs => 80B pitch: 16B aligned, 20-dword bank stride

// Cap VGPRs under 256 (no s_set_vgpr_msb) and keep >=4 waves/SIMD resident.
__global__ __launch_bounds__(256, 4) void arnet_wmma_kernel(
    const float* __restrict__ x, const float* __restrict__ y,
    const float* __restrict__ w, float* __restrict__ out, int numTiles)
{
    // Toeplitz W[j,i] = w[j-i] has only 6 unique 32x16 blocks, keyed by
    // d = (j0-i0)/16 + 1 in 0..5:  U[d][i'][j'] = w[16*(d-1) + j' - i'].
    // fp16 hi/lo split for fp32-equivalent accuracy via 3-term WMMA.
    __shared__ f16_t Uh[6][16][U_PITCH];
    __shared__ f16_t Ul[6][16][U_PITCH];

    for (int idx = threadIdx.x; idx < 6 * 16 * 32; idx += blockDim.x) {
        int dd = idx >> 9, rem = idx & 511, i = rem >> 5, j = rem & 31;
        int kk = 16 * (dd - 1) + j - i;
        float v = (kk >= 0 && kk < 64) ? w[kk] : 0.0f;
        f16_t h = (f16_t)v;
        Uh[dd][i][j] = h;
        Ul[dd][i][j] = (f16_t)(v - (float)h);
    }
    __syncthreads();

    const int lane  = threadIdx.x & 31;
    const int m     = lane & 15;
    const int hi8   = (lane >> 4) << 3;   // 0 / 8  (A-matrix K sub-block select)
    const int jc    = (lane >> 4) << 4;   // 0 / 16 (B-matrix K half select)
    const int wave  = blockIdx.x * (blockDim.x >> 5) + (threadIdx.x >> 5);
    const int nWav  = gridDim.x * (blockDim.x >> 5);

    // Preload the 12 unique B fragments (hi/lo) once; loop-invariant, ~96 VGPRs.
    // B fragment, 16-bit layout: lane = column i'=m; elems 0..7 ->
    // K-row j' = hi8 + e, elems 8..15 -> j' = hi8 + 16 + (e-8).
    Frag16 Bh[6], Bl[6];
#pragma unroll
    for (int d = 0; d < 6; ++d) {
        const f16_t* ph = &Uh[d][m][hi8];
        const f16_t* pl = &Ul[d][m][hi8];
        Bh[d].h[0] = *(const v8h*)(ph);
        Bh[d].h[1] = *(const v8h*)(ph + 16);
        Bl[d].h[0] = *(const v8h*)(pl);
        Bl[d].h[1] = *(const v8h*)(pl + 16);
    }

#pragma clang loop unroll(disable)
    for (int t = wave; t < numTiles; t += nWav) {
        const long long b0 = (long long)t * 16;
        const float* px = x + (b0 + m) * 64 + hi8;
        const float* py = y + (b0 + m) * 64 + hi8;

        v8f acc0 = {}, acc1 = {}, acc2 = {}, acc3 = {};

        // A fragment, 16-bit layout: lane m = batch row; elems 0..7 -> K=j0..j0+7,
        // elems 8..15 -> K=j0+16..j0+23, j0 = 32k + (lane>=16 ? 8 : 0).
#pragma unroll
        for (int k = 0; k < 4; ++k) {
            const float* p = (k < 2) ? (px + 32 * k) : (py + 32 * (k - 2));
            v4f c0 = *(const v4f*)(p);
            v4f c1 = *(const v4f*)(p + 4);
            v4f c2 = *(const v4f*)(p + 16);
            v4f c3 = *(const v4f*)(p + 20);
            float f[16];
#pragma unroll
            for (int e = 0; e < 4; ++e) {
                f[e] = c0[e]; f[4 + e] = c1[e]; f[8 + e] = c2[e]; f[12 + e] = c3[e];
            }
            Frag16 ah, al;
#pragma unroll
            for (int e = 0; e < 16; ++e) {
                f16_t h = (f16_t)f[e];
                ah.v[e] = h;
                al.v[e] = (f16_t)(f[e] - (float)h);
            }

#pragma unroll
            for (int n = 0; n < 4; ++n) {
                if ((k == 0 && n >= 2) || (k == 3 && n <= 1)) continue; // zero band blocks
                const int d = 2 * k - n + 1;                            // unique block id

                v8f a = (n == 0) ? acc0 : (n == 1) ? acc1 : (n == 2) ? acc2 : acc3;
                a = __builtin_amdgcn_wmma_f32_16x16x32_f16(
                        false, ah.v, false, Bh[d].v, (short)0, a, false, false);
                a = __builtin_amdgcn_wmma_f32_16x16x32_f16(
                        false, ah.v, false, Bl[d].v, (short)0, a, false, false);
                a = __builtin_amdgcn_wmma_f32_16x16x32_f16(
                        false, al.v, false, Bh[d].v, (short)0, a, false, false);
                if (n == 0) acc0 = a; else if (n == 1) acc1 = a;
                else if (n == 2) acc2 = a; else acc3 = a;
            }
        }

        // D layout: VGPR r -> M=r (lanes 0-15) / M=r+8 (lanes 16-31), N = lane&15
        float* po = out + (b0 + hi8) * 64 + m;
#pragma unroll
        for (int r = 0; r < 8; ++r) po[r * 64 +  0] = acc0[r];
#pragma unroll
        for (int r = 0; r < 8; ++r) po[r * 64 + 16] = acc1[r];
#pragma unroll
        for (int r = 0; r < 8; ++r) po[r * 64 + 32] = acc2[r];
#pragma unroll
        for (int r = 0; r < 8; ++r) po[r * 64 + 48] = acc3[r];
    }
}

// Scalar fallback for a batch remainder that is not a multiple of 16 (unused at B=1e6).
__global__ void arnet_tail_kernel(const float* __restrict__ x, const float* __restrict__ y,
                                  const float* __restrict__ w, float* __restrict__ out,
                                  long long startRow, long long nRows)
{
    long long e = (long long)blockIdx.x * blockDim.x + threadIdx.x;
    long long total = nRows * 64;
    if (e >= total) return;
    long long b = startRow + (e >> 6);
    int i = (int)(e & 63);
    float s = 0.0f;
    for (int k = 0; k < 64; ++k) {
        int j = i + k;
        float v = (j < 64) ? x[b * 64 + j] : y[b * 64 + (j - 64)];
        s += v * w[k];
    }
    out[b * 64 + i] = s;
}

extern "C" void kernel_launch(void* const* d_in, const int* in_sizes, int n_in,
                              void* d_out, int out_size, void* d_ws, size_t ws_size,
                              hipStream_t stream)
{
    const float* x = (const float*)d_in[0];
    const float* y = (const float*)d_in[1];  // true_output
    const float* w = (const float*)d_in[2];
    float* out = (float*)d_out;

    const long long B = (long long)in_sizes[0] / 64;
    const int numTiles = (int)(B / 16);
    const long long rem = B - (long long)numTiles * 16;

    if (numTiles > 0) {
        int blocks = (numTiles + 7) / 8;
        if (blocks > 2048) blocks = 2048;
        arnet_wmma_kernel<<<blocks, 256, 0, stream>>>(x, y, w, out, numTiles);
    }
    if (rem > 0) {
        long long elems = rem * 64;
        int tb = (int)((elems + 255) / 256);
        arnet_tail_kernel<<<tb, 256, 0, stream>>>(x, y, w, out,
                                                  (long long)numTiles * 16, rem);
    }
}
